// DistanceBiasedFastAttention_23811298689320
// MI455X (gfx1250) — matrix operand order repaired and checked by reading, not compiled, not run on hardware
//
#include <hip/hip_runtime.h>
#include <hip/hip_bf16.h>
#include <math.h>

// Problem constants (from reference setup_inputs)
#define BB   4
#define LL   4096
#define HH   8
#define DD   64
#define DVV  64
#define MM   256          // random features
#define DCAT 72           // D + S/H = 64 + 8
#define EPAD 80           // e-dim 65 padded to 5x16
#define TWO_PI_F 6.283185307179586f
#define EPS_F 1e-6f
#define NORMALIZER 0.35355339059327373f   // 64^-0.25
#define RSQRT_M 0.0625f                    // 1/sqrt(256)

typedef __attribute__((ext_vector_type(16))) __bf16 v16bf;
typedef __attribute__((ext_vector_type(8)))  float  v8f;

union FragU {
  v16bf v;
  unsigned short u[16];
  uint4 q[2];
};

// Round-to-nearest-even f32->bf16 (used in proj packing, A-fragment setup and the ones column).
__device__ __forceinline__ unsigned short f2bf(float f) {
  union { float f; unsigned u; } c; c.f = f;
  unsigned r = c.u + 0x7FFFu + ((c.u >> 16) & 1u);
  return (unsigned short)(r >> 16);
}

// Truncating f32->bf16 for hot paths: compiles to a single high-half move.
__device__ __forceinline__ unsigned short f2bf_fast(float f) {
  union { float f; unsigned u; } c; c.f = f;
  return (unsigned short)(c.u >> 16);
}

// Wave-scope LDS ordering: per-wave private LDS regions only need same-wave
// store->load ordering (DS pipe is in-order per wave); the fence+wave_barrier
// stops the compiler reordering and forces the dscnt wait, with no s_barrier.
__device__ __forceinline__ void wave_sync() {
  __builtin_amdgcn_fence(__ATOMIC_RELEASE, "wavefront");
  __builtin_amdgcn_wave_barrier();
  __builtin_amdgcn_fence(__ATOMIC_ACQUIRE, "wavefront");
}

// ---------------------------------------------------------------------------
// Kernel 0: pre-pack qk_proj (256 x 72 f32) into bf16 WMMA B-fragment tiles.
// Tile (nt 0..15, kt 0..2): element tt of lane l = proj[nt*16+(l&15)]
//                                                   [kt*32+((l<16)?0:16)+tt]
// 48 tiles x 512 halfs = 48 KB.
// ---------------------------------------------------------------------------
__global__ void __launch_bounds__(256)
proj_pack_kernel(const float* __restrict__ proj, unsigned short* __restrict__ pb)
{
  int idx = blockIdx.x * 256 + threadIdx.x;       // [0, 48*512)
  int tile = idx >> 9, r = idx & 511;
  int lane = r >> 4, tt = r & 15;
  int nt = tile / 3, kt = tile - nt * 3;
  int n = nt * 16 + (lane & 15);
  int k = kt * 32 + ((lane < 16) ? 0 : 16) + tt;
  float v = (k < DCAT) ? proj[(size_t)n * DCAT + k] : 0.0f;
  pb[idx] = f2bf(v);
}

// ---------------------------------------------------------------------------
// Kernel 1: feature map  phi(concat(x*norm, fourier))  ->  bf16, WMMA-tiled.
//   is_q=1: output tiled as (seq x m) A-fragments  [tile = rt*8 + kt]
//   is_q=0: output tiled as (m x seq) A-fragments  [tile = mt*128 + st]
// One wave = 16 seq rows x all 256 features; (16x96)x(96x256) in 3 bf16
// WMMA k-steps (K=72 zero-padded to 96). proj fragments are staged once per
// block into LDS so the hot loop's B-operand comes from ds_load_b128.
// ---------------------------------------------------------------------------
__global__ void __launch_bounds__(256)
feature_kernel(const float* __restrict__ x, const float* __restrict__ xs,
               const float* __restrict__ W, const unsigned short* __restrict__ pb,
               const float* __restrict__ a, unsigned short* __restrict__ outp,
               int is_q)
{
  __shared__ unsigned short pj[48 * 512];  // 48 KB proj fragment table
  __shared__ unsigned short lds[8][256];   // per-wave 16x16 bounce tile
  const int tid  = threadIdx.x;
  const int wid  = tid >> 5, lane = tid & 31;
  const int sid  = blockIdx.x * 8 + wid;   // strip id: [0, B*H*L/16)
  const int rt   = sid & 255;              // seq tile (16 rows)
  const int h    = (sid >> 8) & 7;
  const int b    = sid >> 11;

  // One-time cooperative stage of proj fragments (3072 x uint4)
  for (int idx = tid; idx < 48 * 512 / 8; idx += 256)
    reinterpret_cast<uint4*>(pj)[idx] = reinterpret_cast<const uint4*>(pb)[idx];
  __syncthreads();

  const int   rowA = lane & 15;
  const int   seqA = rt * 16 + rowA;
  const float* xrow = x + (((size_t)b * LL + seqA) * HH + h) * DD;

  // Fourier positional features for this row/head
  const float xsv = xs[(size_t)b * LL + seqA];
  const float ah  = is_q ? a[h] : 1.0f;
  float sp[8];
#pragma unroll
  for (int jj = 0; jj < 8; ++jj) {
    int idx = h * 8 + jj;
    float pv = (idx < 32) ? __sinf(TWO_PI_F * W[idx] * xsv)
                          : __cosf(TWO_PI_F * W[idx - 32] * xsv);
    sp[jj] = ah * pv;
  }

  // h = 0.5 * ||xcat||^2
  float s2 = 0.0f;
  const float4* x4 = reinterpret_cast<const float4*>(xrow);
#pragma unroll
  for (int t = 0; t < 16; ++t) {
    float4 vv = x4[t];
    s2 += vv.x * vv.x + vv.y * vv.y + vv.z * vv.z + vv.w * vv.w;
  }
  s2 *= NORMALIZER * NORMALIZER;
#pragma unroll
  for (int jj = 0; jj < 8; ++jj) s2 += sp[jj] * sp[jj];
  const float hval = 0.5f * s2;

  // Hoisted broadcast of h for the 8 D-tile rows this lane owns
  float hb[8];
#pragma unroll
  for (int i = 0; i < 8; ++i)
    hb[i] = __shfl(hval, i + ((lane >= 16) ? 8 : 0), 32);

  // A fragments (K padded 72 -> 96), built once per strip
  FragU afr[3];
#pragma unroll
  for (int kt = 0; kt < 3; ++kt) {
    int kbase = kt * 32 + ((lane < 16) ? 0 : 16);
#pragma unroll
    for (int half = 0; half < 2; ++half) {
      int k0 = kbase + half * 8;
#pragma unroll
      for (int t = 0; t < 8; ++t) {
        int k = k0 + t;
        float v = (k < DD)   ? xrow[k] * NORMALIZER
                : (k < DCAT) ? sp[k - DD]
                : 0.0f;
        afr[kt].u[half * 8 + t] = f2bf(v);
      }
    }
  }

  const size_t plane = (size_t)(b * HH + h) * LL * MM;

  for (int nt = 0; nt < 16; ++nt) {
    v8f acc = {0.f, 0.f, 0.f, 0.f, 0.f, 0.f, 0.f, 0.f};
#pragma unroll
    for (int kt = 0; kt < 3; ++kt) {
      FragU bfr;
      const unsigned short* bp = &pj[((nt * 3 + kt) << 9) + lane * 16];
      bfr.q[0] = reinterpret_cast<const uint4*>(bp)[0];
      bfr.q[1] = reinterpret_cast<const uint4*>(bp)[1];
      acc = __builtin_amdgcn_wmma_f32_16x16x32_bf16(
          false, afr[kt].v, false, bfr.v, (short)0, acc, false, false);
    }
    // phi = exp(acc - h) / sqrt(M); bounce through per-wave LDS region
#pragma unroll
    for (int i = 0; i < 8; ++i) {
      int row = i + ((lane >= 16) ? 8 : 0);           // D-tile row for VGPR i
      unsigned short dbf = f2bf_fast(__expf(acc[i] - hb[i]) * RSQRT_M);
      int addr = is_q ? (row * 16 + (lane & 15))      // row-major (seq, m)
                      : ((lane & 15) * 16 + row);     // m-major  (m, seq)
      lds[wid][addr] = dbf;
    }
    wave_sync();
    const int rbase = (lane & 15) * 16 + ((lane >= 16) ? 8 : 0);
    uint4 pk = *reinterpret_cast<const uint4*>(&lds[wid][rbase]);
    size_t tile, off;
    if (is_q) { tile = (size_t)rt * 8 + (nt >> 1);  off = (size_t)(nt & 1) * 8; }
    else      { tile = (size_t)nt * 128 + (rt >> 1); off = (size_t)(rt & 1) * 8; }
    *reinterpret_cast<uint4*>(outp + plane + tile * 512 + (size_t)((lane & 15) + 2 * off) * 16 + ((lane >= 16) ? 8 : 0)) = pk;
    wave_sync();
  }
}

// ---------------------------------------------------------------------------
// Kernel 2: buf1[b,h,m,e] = sum_k ksprime[k,m] * c[k,e],  c = [v | 1 | 0-pad]
// Grid: (b,h) x 4 L-chunks of 1024; partials combined with f32 atomics.
// c-tile staged in LDS *already in B-fragment order* -> fragment = 2x b128.
// ---------------------------------------------------------------------------
__global__ void __launch_bounds__(256)
buf1_kernel(const float* __restrict__ vs, const unsigned short* __restrict__ kprime,
            float* __restrict__ buf1)
{
  __shared__ unsigned short cst[5 * 512];    // 5 et-fragments of 32x16
  const int tid = threadIdx.x, wid = tid >> 5, lane = tid & 31;
  const int chunk = blockIdx.x & 3;
  const int bh = blockIdx.x >> 2;
  const int b = bh >> 3, h = bh & 7;
  const int seqBase = chunk * 1024;
  const size_t kplane = (size_t)bh * LL * MM;

  const v8f zero = {0.f, 0.f, 0.f, 0.f, 0.f, 0.f, 0.f, 0.f};
  v8f accs[10];
#pragma unroll
  for (int j = 0; j < 10; ++j) accs[j] = zero;

  for (int ks = 0; ks < 32; ++ks) {
    {   // stage c-tile (32 seq x 80 e) in fragment order:
        // dest = et*512 + lane*16 + tt, lane = (e&15)+16*(r>=16), tt = r&15
      int r = tid >> 3, cg = tid & 7;
      int sr = seqBase + ks * 32 + r;
      const float* vrow = vs + (((size_t)b * LL + sr) * HH + h) * DVV;
      const float4* v4 = reinterpret_cast<const float4*>(vrow + cg * 8);
      float4 v0 = v4[0], v1 = v4[1];
      float vals[8] = {v0.x, v0.y, v0.z, v0.w, v1.x, v1.y, v1.z, v1.w};
      const int lgrp = (r >= 16) ? 16 : 0;
      const int tt = r & 15;
#pragma unroll
      for (int t = 0; t < 8; ++t) {
        int e = cg * 8 + t;
        int dest = ((e >> 4) << 9) + ((e & 15) + lgrp) * 16 + tt;
        cst[dest] = f2bf_fast(vals[t]);
      }
      if (cg == 7) {                       // e = 64 (ones) .. 79 (pad)
#pragma unroll
        for (int e = 64; e < EPAD; ++e) {
          int dest = (4 << 9) + ((e & 15) + lgrp) * 16 + tt;
          cst[dest] = (e == 64) ? f2bf(1.0f) : (unsigned short)0;
        }
      }
    }
    __syncthreads();
    const int st = (seqBase >> 5) + ks;
#pragma unroll
    for (int j = 0; j < 10; ++j) {
      const int t  = wid + 8 * j;
      const int mt = t & 15, et = t >> 4;
      FragU af;
      const unsigned short* ap =
          kprime + kplane + ((size_t)mt * 128 + st) * 512 + (size_t)lane * 16;
      af.q[0] = reinterpret_cast<const uint4*>(ap)[0];
      af.q[1] = reinterpret_cast<const uint4*>(ap)[1];
      __builtin_prefetch(ap + 512, 0, 1);
      FragU bf_;
      const unsigned short* bp = &cst[(et << 9) + lane * 16];
      bf_.q[0] = reinterpret_cast<const uint4*>(bp)[0];
      bf_.q[1] = reinterpret_cast<const uint4*>(bp)[1];
      accs[j] = __builtin_amdgcn_wmma_f32_16x16x32_bf16(
          false, af.v, false, bf_.v, (short)0, accs[j], false, false);
    }
    __syncthreads();
  }
#pragma unroll
  for (int j = 0; j < 10; ++j) {
    const int t  = wid + 8 * j;
    const int mt = t & 15, et = t >> 4;
    const int col = et * 16 + (lane & 15);
#pragma unroll
    for (int i = 0; i < 8; ++i) {
      int m = mt * 16 + i + ((lane >= 16) ? 8 : 0);
      unsafeAtomicAdd(&buf1[(size_t)bh * MM * EPAD + (size_t)m * EPAD + col],
                      accs[j][i]);
    }
  }
}

// ---------------------------------------------------------------------------
// Kernel 3: buf2 = qsprime @ buf1, then out = num / max(den, EPS).
// One WG per (b,h, 128-seq chunk); the 256x80 buf1 plane is staged in LDS
// pre-swizzled into the 40 (kt x et) B-fragment tiles -> fragment = 2x b128.
// ---------------------------------------------------------------------------
__global__ void __launch_bounds__(256)
out_kernel(const unsigned short* __restrict__ qprime, const float* __restrict__ buf1,
           float* __restrict__ out)
{
  __shared__ unsigned short b1s[40 * 512];   // 40 KB bf16, fragment order
  const int tid = threadIdx.x, wid = tid >> 5, lane = tid & 31;
  const int chunk = blockIdx.x & 31;
  const int bh = blockIdx.x >> 5;
  const int b = bh >> 3, h = bh & 7;
  const size_t qplane = (size_t)bh * LL * MM;

  for (int idx = tid; idx < MM * EPAD; idx += 256) {
    int m = idx / EPAD, e = idx - m * EPAD;
    int kt = m >> 5, mi = m & 31;
    int dl = (e & 15) + ((mi >= 16) ? 16 : 0);
    int dest = ((kt * 5 + (e >> 4)) << 9) + dl * 16 + (mi & 15);
    b1s[dest] = f2bf_fast(buf1[(size_t)bh * MM * EPAD + idx]);
  }
  __syncthreads();

  const int rt = chunk * 8 + wid;   // 16-row seq tile per wave
  const v8f zero = {0.f, 0.f, 0.f, 0.f, 0.f, 0.f, 0.f, 0.f};
  v8f accs[5];
#pragma unroll
  for (int et = 0; et < 5; ++et) accs[et] = zero;

#pragma unroll
  for (int kt = 0; kt < 8; ++kt) {           // m contraction: 8 x K=32
    FragU af;
    const unsigned short* ap =
        qprime + qplane + ((size_t)rt * 8 + kt) * 512 + (size_t)lane * 16;
    af.q[0] = reinterpret_cast<const uint4*>(ap)[0];
    af.q[1] = reinterpret_cast<const uint4*>(ap)[1];
#pragma unroll
    for (int et = 0; et < 5; ++et) {
      FragU bf_;
      const unsigned short* bp = &b1s[((kt * 5 + et) << 9) + lane * 16];
      bf_.q[0] = reinterpret_cast<const uint4*>(bp)[0];
      bf_.q[1] = reinterpret_cast<const uint4*>(bp)[1];
      accs[et] = __builtin_amdgcn_wmma_f32_16x16x32_bf16(
          false, af.v, false, bf_.v, (short)0, accs[et], false, false);
    }
  }

  // finalize: den lives at e=64 -> tile 4, col 0 -> lanes 0 / 16
#pragma unroll
  for (int i = 0; i < 8; ++i) {
    float den = __shfl(accs[4][i], (lane < 16) ? 0 : 16, 32);
    den = (den < EPS_F) ? EPS_F : den;
    const float rden = 1.0f / den;
    const int row  = i + ((lane >= 16) ? 8 : 0);
    const int seqr = rt * 16 + row;
    const size_t obase = (((size_t)b * LL + seqr) * HH + h) * 64;
#pragma unroll
    for (int et = 0; et < 4; ++et)
      out[obase + et * 16 + (lane & 15)] = accs[et][i] * rden;
  }
}

// ---------------------------------------------------------------------------
extern "C" void kernel_launch(void* const* d_in, const int* in_sizes, int n_in,
                              void* d_out, int out_size, void* d_ws, size_t ws_size,
                              hipStream_t stream) {
  const float* qs     = (const float*)d_in[0];
  const float* ks     = (const float*)d_in[1];
  const float* vs     = (const float*)d_in[2];
  const float* qs_s   = (const float*)d_in[3];
  const float* ks_s   = (const float*)d_in[4];
  const float* four_W = (const float*)d_in[5];
  const float* proj   = (const float*)d_in[6];
  const float* a      = (const float*)d_in[7];

  const size_t featElems = (size_t)BB * HH * LL * MM;      // bf16 elements
  unsigned short* qprime = (unsigned short*)d_ws;
  unsigned short* kprime = qprime + featElems;
  float* buf1 = (float*)(kprime + featElems);
  const size_t buf1Elems = (size_t)BB * HH * MM * EPAD;
  unsigned short* projb = (unsigned short*)(buf1 + buf1Elems);

  hipMemsetAsync(buf1, 0, buf1Elems * sizeof(float), stream);

  // 48 tiles x 512 halfs of bf16-packed proj fragments
  proj_pack_kernel<<<dim3(96), dim3(256), 0, stream>>>(proj, projb);

  // B*H*L/16 strips / 8 waves per block = 1024 blocks
  feature_kernel<<<dim3(1024), dim3(256), 0, stream>>>(qs, qs_s, four_W, projb, a, qprime, 1);
  feature_kernel<<<dim3(1024), dim3(256), 0, stream>>>(ks, ks_s, four_W, projb, a, kprime, 0);
  // (b,h) x 4 seq-chunks
  buf1_kernel<<<dim3(128), dim3(256), 0, stream>>>(vs, kprime, buf1);
  // (b,h) x 32 seq-chunks of 128 rows
  out_kernel<<<dim3(1024), dim3(256), 0, stream>>>(qprime, buf1, (float*)d_out);
}
